// LohaModule_3753801417310
// MI455X (gfx1250) — compile-verified, matrix operand order checked by reading, not run on hardware
//
#include <hip/hip_runtime.h>

// ---------- problem constants ----------
#define MDIM  16384      // BATCH*SEQ
#define NDIM  4096       // OUT_DIM
#define KDIM  4096       // IN_DIM
#define RANK  32
#define SCALE 0.5f       // ALPHA/RANK * MULTIPLIER * SCALAR

// ---------- vector types ----------
typedef __attribute__((ext_vector_type(16))) __bf16        bf16x16;
typedef __attribute__((ext_vector_type(8)))  float         floatx8;
typedef __attribute__((ext_vector_type(4)))  float         floatx4;
typedef __attribute__((ext_vector_type(4)))  unsigned int  uintx4;
typedef __attribute__((ext_vector_type(2)))  unsigned int  uintx2;

union Frag {
    bf16x16 v;
    uintx4  q[2];
};

// round-to-nearest-even f32 -> bf16, packed x2
static __device__ inline unsigned int cvt2_bf16(float a, float b) {
    unsigned int ua = __float_as_uint(a);
    unsigned int ub = __float_as_uint(b);
    ua = (ua + 0x7FFFu + ((ua >> 16) & 1u)) >> 16;
    ub = (ub + 0x7FFFu + ((ub >> 16) & 1u)) >> 16;
    return (ua & 0xFFFFu) | (ub << 16);
}

// ============================================================
// Kernel 1: W_merged = org + SCALE * (w1a@w1b) . (w2a@w2b), stored as bf16
// 64(out) x 64(in) tile per 256-thread block.
// ============================================================
__global__ __launch_bounds__(256) void merge_weight_kernel(
    const float* __restrict__ orgw,
    const float* __restrict__ w1a, const float* __restrict__ w1b,
    const float* __restrict__ w2a, const float* __restrict__ w2b,
    unsigned short* __restrict__ Wbf)
{
    __shared__ float A1[64 * RANK];   // w1a rows  [o][r]
    __shared__ float A2[64 * RANK];
    __shared__ float B1[RANK * 64];   // w1b cols  [r][i]
    __shared__ float B2[RANK * 64];

    const int i0 = blockIdx.x * 64;
    const int o0 = blockIdx.y * 64;
    const int t  = threadIdx.x;

    // w1a/w2a rows o0..o0+63 are one contiguous 2048-float block
    #pragma unroll
    for (int l = t; l < 64 * RANK; l += 256) {
        A1[l] = w1a[o0 * RANK + l];
        A2[l] = w2a[o0 * RANK + l];
        int r = l >> 6, c = l & 63;
        B1[l] = w1b[r * KDIM + i0 + c];
        B2[l] = w2b[r * KDIM + i0 + c];
    }
    __syncthreads();

    const int to = t >> 4;     // 0..15 -> 4 output rows each
    const int ti = t & 15;     // 0..15 -> 4 input cols each

    float s1[4][4] = {};
    float s2[4][4] = {};
    #pragma unroll 4
    for (int r = 0; r < RANK; ++r) {
        float a1[4], a2[4], b1[4], b2[4];
        #pragma unroll
        for (int k = 0; k < 4; ++k) {
            a1[k] = A1[(to * 4 + k) * RANK + r];
            a2[k] = A2[(to * 4 + k) * RANK + r];
            b1[k] = B1[r * 64 + ti * 4 + k];
            b2[k] = B2[r * 64 + ti * 4 + k];
        }
        #pragma unroll
        for (int oo = 0; oo < 4; ++oo)
            #pragma unroll
            for (int ii = 0; ii < 4; ++ii) {
                s1[oo][ii] = __builtin_fmaf(a1[oo], b1[ii], s1[oo][ii]);
                s2[oo][ii] = __builtin_fmaf(a2[oo], b2[ii], s2[oo][ii]);
            }
    }

    #pragma unroll
    for (int oo = 0; oo < 4; ++oo) {
        const int o = o0 + to * 4 + oo;
        const int i = i0 + ti * 4;
        floatx4 org = *(const floatx4*)&orgw[(size_t)o * KDIM + i];
        float w0 = org[0] + SCALE * s1[oo][0] * s2[oo][0];
        float w1 = org[1] + SCALE * s1[oo][1] * s2[oo][1];
        float w2 = org[2] + SCALE * s1[oo][2] * s2[oo][2];
        float w3 = org[3] + SCALE * s1[oo][3] * s2[oo][3];
        uintx2 p;
        p[0] = cvt2_bf16(w0, w1);
        p[1] = cvt2_bf16(w2, w3);
        *(uintx2*)&Wbf[(size_t)o * KDIM + i] = p;
    }
}

// ============================================================
// Kernel 2: out = x @ W^T + bias, bf16 WMMA with f32 accumulate.
// Block: 128 threads = 4 waves (2x2). Block tile 128x128, BK=32.
// Wave tile 64x64 -> 16 x v_wmma_f32_16x16x32_bf16 per K-step.
// ============================================================
#define BM  128
#define BN  128
#define BK  32
#define LDT 40     // LDS row stride in bf16 elems (80B = 20 dwords, conflict-free)

__global__ __launch_bounds__(128) void loha_gemm_kernel(
    const float* __restrict__ x,
    const unsigned short* __restrict__ Wbf,
    const float* __restrict__ bias,
    float* __restrict__ out)
{
    __shared__ unsigned short As[2][BM * LDT];   // x tile   [m][k] bf16
    __shared__ unsigned short Bs[2][BN * LDT];   // W tile   [n][k] bf16

    const int t     = threadIdx.x;
    const int wave  = t >> 5;
    const int lane  = t & 31;
    const int waveM = wave >> 1;      // 0..1
    const int waveN = wave & 1;       // 0..1
    const int h     = lane >> 4;      // k-half selector (WMMA lane layout)
    const int idx   = lane & 15;      // m (A frag) / n (B frag)

    const int gRow0 = blockIdx.y * BM;
    const int gCol0 = blockIdx.x * BN;

    // loader thread mapping
    const int ar = t >> 3;  // 0..15  (x rows, step 16)
    const int ak = t & 7;   // 0..7   (float4 column group)
    const int br = t >> 2;  // 0..31  (W rows, step 32)
    const int bk = t & 3;   // 0..3   (uint4 = 8 bf16 column group)

    floatx4 xreg[8];
    uintx4  wreg[4];

    auto load_global = [&](int kt) {
        const float* xp = x + (size_t)gRow0 * KDIM + kt * BK;
        #pragma unroll
        for (int i2 = 0; i2 < 8; ++i2) {
            int row = ar + 16 * i2;
            xreg[i2] = *(const floatx4*)(xp + (size_t)row * KDIM + ak * 4);
        }
        const unsigned short* wp = Wbf + (size_t)gCol0 * KDIM + kt * BK;
        #pragma unroll
        for (int i2 = 0; i2 < 4; ++i2) {
            int row = br + 32 * i2;
            wreg[i2] = *(const uintx4*)(wp + (size_t)row * KDIM + bk * 8);
        }
    };
    auto store_lds = [&](int buf) {
        #pragma unroll
        for (int i2 = 0; i2 < 8; ++i2) {
            int row = ar + 16 * i2;
            uintx2 p;
            p[0] = cvt2_bf16(xreg[i2][0], xreg[i2][1]);
            p[1] = cvt2_bf16(xreg[i2][2], xreg[i2][3]);
            *(uintx2*)&As[buf][row * LDT + ak * 4] = p;
        }
        #pragma unroll
        for (int i2 = 0; i2 < 4; ++i2) {
            int row = br + 32 * i2;
            *(uintx4*)&Bs[buf][row * LDT + bk * 8] = wreg[i2];
        }
    };

    floatx8 acc[4][4] = {};

    load_global(0);
    store_lds(0);
    __syncthreads();

    const int nk = KDIM / BK;  // 128
    for (int kt = 0; kt < nk; ++kt) {
        const int cur = kt & 1;
        if (kt + 1 < nk) load_global(kt + 1);   // overlap global latency with WMMA

        // A fragments: lane holds row m=idx, k in [h*8, h*8+8) U [16+h*8, 16+h*8+8)
        Frag afrag[4], bfrag[4];
        #pragma unroll
        for (int mi = 0; mi < 4; ++mi) {
            int row = waveM * 64 + mi * 16 + idx;
            afrag[mi].q[0] = *(const uintx4*)&As[cur][row * LDT + h * 8];
            afrag[mi].q[1] = *(const uintx4*)&As[cur][row * LDT + 16 + h * 8];
        }
        #pragma unroll
        for (int ni = 0; ni < 4; ++ni) {
            int row = waveN * 64 + ni * 16 + idx;
            bfrag[ni].q[0] = *(const uintx4*)&Bs[cur][row * LDT + h * 8];
            bfrag[ni].q[1] = *(const uintx4*)&Bs[cur][row * LDT + 16 + h * 8];
        }

        #pragma unroll
        for (int mi = 0; mi < 4; ++mi)
            #pragma unroll
            for (int ni = 0; ni < 4; ++ni)
                acc[mi][ni] = __builtin_amdgcn_wmma_f32_16x16x32_bf16(
                    /*neg_a=*/false, afrag[mi].v,
                    /*neg_b=*/false, bfrag[ni].v,
                    /*c_mod=*/(short)0, acc[mi][ni],
                    /*reuse_a=*/false, /*reuse_b=*/false);

        if (kt + 1 < nk) store_lds((kt + 1) & 1);
        __syncthreads();
    }

    // Epilogue: C layout — lanes 0-15: M=j, lanes 16-31: M=j+8; N = idx
    #pragma unroll
    for (int ni = 0; ni < 4; ++ni) {
        const int col = gCol0 + waveN * 64 + ni * 16 + idx;
        const float bv = bias[col];
        #pragma unroll
        for (int mi = 0; mi < 4; ++mi) {
            const int rbase = gRow0 + waveM * 64 + mi * 16 + 8 * h;
            #pragma unroll
            for (int j = 0; j < 8; ++j) {
                out[(size_t)(rbase + j) * NDIM + col] = acc[mi][ni][j] + bv;
            }
        }
    }
}

// ============================================================
extern "C" void kernel_launch(void* const* d_in, const int* in_sizes, int n_in,
                              void* d_out, int out_size, void* d_ws, size_t ws_size,
                              hipStream_t stream)
{
    (void)in_sizes; (void)n_in; (void)out_size; (void)ws_size;

    const float* x    = (const float*)d_in[0];
    const float* orgw = (const float*)d_in[1];
    const float* orgb = (const float*)d_in[2];
    const float* w1a  = (const float*)d_in[3];
    const float* w1b  = (const float*)d_in[4];
    const float* w2a  = (const float*)d_in[5];
    const float* w2b  = (const float*)d_in[6];
    float* out = (float*)d_out;

    unsigned short* Wbf = (unsigned short*)d_ws;   // 4096*4096 bf16 = 32 MB

    dim3 mgrid(KDIM / 64, NDIM / 64);              // 64 x 64 blocks
    merge_weight_kernel<<<mgrid, 256, 0, stream>>>(orgw, w1a, w1b, w2a, w2b, Wbf);

    dim3 ggrid(NDIM / BN, MDIM / BM);              // 32 x 128 blocks, N fastest for L2 reuse
    loha_gemm_kernel<<<ggrid, 128, 0, stream>>>(x, Wbf, orgb, out);
}
